// GAT_13400297963989
// MI455X (gfx1250) — matrix operand order repaired and checked by reading, not compiled, not run on hardware
//
#include <hip/hip_runtime.h>
#include <stdint.h>

#define NN   50000
#define DIN  256
#define HH   8
#define CC   32
#define OUTF 32
#define EE   800000
#define ET   (EE + NN)
#define HID  256   // HH*CC

typedef __attribute__((ext_vector_type(16))) __bf16   v16bf;
typedef __attribute__((ext_vector_type(8)))  float    v8f;
typedef __attribute__((ext_vector_type(4)))  unsigned u32x4;

union Frag16 { u32x4 q[2]; v16bf v; };

static __device__ __forceinline__ unsigned short f2bf(float f) {
    unsigned u = __float_as_uint(f);
    unsigned r = u + 0x7FFFu + ((u >> 16) & 1u);   // round to nearest even
    return (unsigned short)(r >> 16);
}
// order-preserving uint encoding of float (for atomicMax-based segment max)
static __device__ __forceinline__ unsigned fkey(float f) {
    unsigned u = __float_as_uint(f);
    return (u & 0x80000000u) ? ~u : (u | 0x80000000u);
}
static __device__ __forceinline__ float fkey_inv(unsigned k) {
    unsigned u = (k & 0x80000000u) ? (k & 0x7FFFFFFFu) : ~k;
    return __uint_as_float(u);
}

__global__ void k_fill_u32(unsigned* __restrict__ p, long n, unsigned v) {
    long i = (long)blockIdx.x * blockDim.x + threadIdx.x;
    if (i < n) p[i] = v;
}

__global__ void k_cvt_bf16(const float* __restrict__ x, unsigned short* __restrict__ y, long n) {
    long i = (long)blockIdx.x * blockDim.x + threadIdx.x;
    if (i < n) y[i] = f2bf(x[i]);
}

// Pack row-major W[K x NC] (f32) into WMMA B-operand order (bf16):
// P[((kb*nct + nb)*32 + lane)*16 + idx], lane = (k%32)/16*16 + n%16, idx = k%16.
__global__ void k_pack_b(const float* __restrict__ W, unsigned short* __restrict__ P,
                         int K, int NC) {
    int i = blockIdx.x * blockDim.x + threadIdx.x;
    if (i >= K * NC) return;
    int k = i / NC, n = i - k * NC;
    int kb = k >> 5, kr = k & 31, nb = n >> 4, nr = n & 15;
    int nct = NC >> 4;
    int lane = ((kr >> 4) << 4) | nr;
    int idx  = kr & 15;
    P[((size_t)((kb * nct + nb) * 32 + lane) << 4) + idx] = f2bf(W[(size_t)k * NC + n]);
}

// C[rows x NC] = A[rows x K](bf16, row-major) @ Bpacked. One wave -> one 16x16 tile.
__global__ void k_wmma_gemm(const unsigned short* __restrict__ A,
                            const unsigned short* __restrict__ Bp,
                            float* __restrict__ C, int K, int NC) {
    const int lane = threadIdx.x;            // 0..31
    const int lo = lane & 15, hi = lane >> 4;
    const int nct = NC >> 4;
    const int colTile = blockIdx.y * blockDim.y + threadIdx.y;   // wave-uniform
    if (colTile >= nct) return;
    const long rowBase = (long)blockIdx.x << 4;
    const unsigned short* arow = A + (size_t)(rowBase + lo) * K;
    const unsigned short* bcol = Bp + ((size_t)(colTile * 32 + lane) << 4);
    const size_t bstep = (size_t)nct << 9;   // elements per K-slab of 32
    v8f acc = {};
    const int nkb = K >> 5;
    for (int kb = 0; kb < nkb; ++kb) {
        Frag16 a, b;
        const unsigned short* ap = arow + (kb << 5) + (hi << 4);
        a.q[0] = *(const u32x4*)(ap);         // K = 32*kb + 16*hi + [0..7]
        a.q[1] = *(const u32x4*)(ap + 8);     // K = 32*kb + 16*hi + 8 + [0..7]
        const unsigned short* bp = bcol + (size_t)kb * bstep;
        b.q[0] = *(const u32x4*)(bp);
        b.q[1] = *(const u32x4*)(bp + 8);
        acc = __builtin_amdgcn_wmma_f32_16x16x32_bf16(false, a.v, false, b.v,
                                                      (short)0, acc, false, false);
    }
    float* crow = C + (size_t)(rowBase + (hi << 3)) * NC + colTile * 16 + lo;
#pragma unroll
    for (int v = 0; v < 8; ++v) crow[(size_t)v * NC] = acc[v];
}

// Per-(node,head): als = <h[n,h,:], a_src[h,:]>, ald likewise. C fixed at 32.
__global__ void k_att_dots(const float* __restrict__ h, const float* __restrict__ asrc,
                           const float* __restrict__ adst, float* __restrict__ als,
                           float* __restrict__ ald, long nNodes, int hshift) {
    long i = (long)blockIdx.x * blockDim.x + threadIdx.x;
    if (i >= (nNodes << hshift)) return;
    long node = i >> hshift;
    int  hh   = (int)(i & ((1 << hshift) - 1));
    const float* hp = h + ((node << hshift) + hh) * CC;
    const float* as = asrc + hh * CC;
    const float* ad = adst + hh * CC;
    float s0 = 0.f, s1 = 0.f;
#pragma unroll
    for (int c = 0; c < CC; ++c) { float t = hp[c]; s0 += t * as[c]; s1 += t * ad[c]; }
    als[i] = s0; ald[i] = s1;
}

static __device__ __forceinline__ void edge_sd(const int* __restrict__ ei, long e,
                                               int& s, int& d) {
    if (e < EE) { s = ei[e]; d = ei[EE + e]; }
    else        { s = d = (int)(e - EE); }     // appended self-loops
}

__global__ void k_edge_logit_max(const int* __restrict__ ei, const float* __restrict__ als,
                                 const float* __restrict__ ald, float* __restrict__ elog,
                                 unsigned* __restrict__ m, int hshift) {
    long i = (long)blockIdx.x * blockDim.x + threadIdx.x;
    if (i >= ((long)ET << hshift)) return;
    long e = i >> hshift;
    int  hh = (int)(i & ((1 << hshift) - 1));
    int s, d; edge_sd(ei, e, s, d);
    float v = als[((long)s << hshift) + hh] + ald[((long)d << hshift) + hh];
    v = v > 0.f ? v : 0.2f * v;                // LeakyReLU(0.2)
    elog[i] = v;
    atomicMax(&m[((long)d << hshift) + hh], fkey(v));
}

__global__ void k_edge_expsum(const int* __restrict__ ei, float* __restrict__ elog,
                              const unsigned* __restrict__ m, float* __restrict__ ssum,
                              int hshift) {
    long i = (long)blockIdx.x * blockDim.x + threadIdx.x;
    if (i >= ((long)ET << hshift)) return;
    long e = i >> hshift;
    int  hh = (int)(i & ((1 << hshift) - 1));
    int s, d; edge_sd(ei, e, s, d);
    float mx = fkey_inv(m[((long)d << hshift) + hh]);
    float ex = __expf(elog[i] - mx);
    elog[i] = ex;
    atomicAdd(&ssum[((long)d << hshift) + hh], ex);
}

// thread per (edge, channel): out[dst,ch] += h[src,ch] * alpha(e, head(ch))
__global__ void k_edge_aggregate(const int* __restrict__ ei, const float* __restrict__ h,
                                 const float* __restrict__ elog, const float* __restrict__ ssum,
                                 float* __restrict__ out, int hshift) {
    int fshift = hshift + 5;                   // F = H*32 channels
    long i = (long)blockIdx.x * blockDim.x + threadIdx.x;
    if (i >= ((long)ET << fshift)) return;
    long e = i >> fshift;
    int ch = (int)(i & ((1 << fshift) - 1));
    int hh = ch >> 5;
    int s, d; edge_sd(ei, e, s, d);
    float alpha = elog[(e << hshift) + hh] / (ssum[((long)d << hshift) + hh] + 1e-16f);
    atomicAdd(&out[((long)d << fshift) + ch], h[((long)s << fshift) + ch] * alpha);
}

__global__ void k_elu_bias_cvt(const float* __restrict__ agg, const float* __restrict__ bias,
                               unsigned short* __restrict__ yb, long n) {
    long i = (long)blockIdx.x * blockDim.x + threadIdx.x;
    if (i >= n) return;
    float v = agg[i] + bias[i & (HID - 1)];
    v = v > 0.f ? v : (__expf(v) - 1.f);       // ELU
    yb[i] = f2bf(v);
}

__global__ void k_add_bias(float* __restrict__ out, const float* __restrict__ b, long n) {
    long i = (long)blockIdx.x * blockDim.x + threadIdx.x;
    if (i < n) out[i] += b[i & (OUTF - 1)];
}

extern "C" void kernel_launch(void* const* d_in, const int* in_sizes, int n_in,
                              void* d_out, int out_size, void* d_ws, size_t ws_size,
                              hipStream_t stream) {
    const float* x     = (const float*)d_in[0];
    const int*   ei    = (const int*)d_in[1];
    // d_in[2] edge_weight unused (edge_dim=None in reference)
    const float* W1    = (const float*)d_in[3];
    const float* asrc1 = (const float*)d_in[4];
    const float* adst1 = (const float*)d_in[5];
    const float* b1    = (const float*)d_in[6];
    const float* W2    = (const float*)d_in[7];
    const float* asrc2 = (const float*)d_in[8];
    const float* adst2 = (const float*)d_in[9];
    const float* b2    = (const float*)d_in[10];
    float* out = (float*)d_out;

    char* ws = (char*)d_ws;
    size_t off = 0;
    auto take = [&](size_t bytes) -> char* {
        char* p = ws + off;
        off = (off + bytes + 255) & ~(size_t)255;
        return p;
    };
    unsigned short* xb  = (unsigned short*)take((size_t)NN * DIN * 2);   // layer1 A; reused as h2b
    unsigned short* w1p = (unsigned short*)take((size_t)DIN * HID * 2);
    unsigned short* w2p = (unsigned short*)take((size_t)HID * OUTF * 2);
    float* h1   = (float*)take((size_t)NN * HID * 4);                    // reused as g2
    float* out1 = (float*)take((size_t)NN * HID * 4);
    float* als  = (float*)take((size_t)NN * HH * 4);                     // reused layer2
    float* ald  = (float*)take((size_t)NN * HH * 4);
    unsigned* mb = (unsigned*)take((size_t)NN * HH * 4);
    float* sb   = (float*)take((size_t)NN * HH * 4);
    float* elog = (float*)take((size_t)ET * HH * 4);
    unsigned short* h2b = xb;
    float* g2 = h1;

    const int B = 256;
    auto g = [](long n) { return (unsigned)((n + 255) / 256); };

    // ---------- layer 1 ----------
    k_cvt_bf16<<<g((long)NN * DIN), B, 0, stream>>>(x, xb, (long)NN * DIN);
    k_pack_b<<<g((long)DIN * HID), B, 0, stream>>>(W1, w1p, DIN, HID);
    k_pack_b<<<g((long)HID * OUTF), B, 0, stream>>>(W2, w2p, HID, OUTF);
    k_wmma_gemm<<<dim3(NN / 16, 2), dim3(32, 8), 0, stream>>>(xb, w1p, h1, DIN, HID);
    k_att_dots<<<g((long)NN * HH), B, 0, stream>>>(h1, asrc1, adst1, als, ald, NN, 3);
    k_fill_u32<<<g((long)NN * HH), B, 0, stream>>>(mb, (long)NN * HH, 0u);
    k_fill_u32<<<g((long)NN * HH), B, 0, stream>>>((unsigned*)sb, (long)NN * HH, 0u);
    k_fill_u32<<<g((long)NN * HID), B, 0, stream>>>((unsigned*)out1, (long)NN * HID, 0u);
    k_edge_logit_max<<<g((long)ET * HH), B, 0, stream>>>(ei, als, ald, elog, mb, 3);
    k_edge_expsum<<<g((long)ET * HH), B, 0, stream>>>(ei, elog, mb, sb, 3);
    k_edge_aggregate<<<g((long)ET * HID), B, 0, stream>>>(ei, h1, elog, sb, out1, 3);
    k_elu_bias_cvt<<<g((long)NN * HID), B, 0, stream>>>(out1, b1, h2b, (long)NN * HID);

    // ---------- layer 2 ----------
    k_wmma_gemm<<<dim3(NN / 16, 1), dim3(32, 2), 0, stream>>>(h2b, w2p, g2, HID, OUTF);
    k_att_dots<<<g((long)NN), B, 0, stream>>>(g2, asrc2, adst2, als, ald, NN, 0);
    k_fill_u32<<<g((long)NN), B, 0, stream>>>(mb, (long)NN, 0u);
    k_fill_u32<<<g((long)NN), B, 0, stream>>>((unsigned*)sb, (long)NN, 0u);
    k_fill_u32<<<g((long)NN * OUTF), B, 0, stream>>>((unsigned*)out, (long)NN * OUTF, 0u);
    k_edge_logit_max<<<g((long)ET), B, 0, stream>>>(ei, als, ald, elog, mb, 0);
    k_edge_expsum<<<g((long)ET), B, 0, stream>>>(ei, elog, mb, sb, 0);
    k_edge_aggregate<<<g((long)ET * OUTF), B, 0, stream>>>(ei, g2, elog, sb, out, 0);
    k_add_bias<<<g((long)NN * OUTF), B, 0, stream>>>(out, b2, (long)NN * OUTF);

    (void)in_sizes; (void)n_in; (void)out_size; (void)ws_size;
}
